// MultiHeadAttention_1666447311339
// MI455X (gfx1250) — matrix operand order repaired and checked by reading, not compiled, not run on hardware
//
#include <hip/hip_runtime.h>
#include <stdint.h>

// Problem constants (match reference)
#define BB   128
#define TT   256
#define DD   384
#define HH   4
#define HS   96
#define VV   65
#define BT   (BB*TT)      // 32768 tokens
#define D3   (3*DD)       // 1152  (qkv concat)
#define DF   (4*DD)       // 1536  (ffn hidden)
#define VPAD 128          // lm head N padded to multiple of 64

typedef unsigned short ushort_t;
typedef __attribute__((ext_vector_type(16))) __bf16 v16bf;
typedef __attribute__((ext_vector_type(8)))  float  v8f;
typedef __attribute__((ext_vector_type(4)))  float  f32x4;

// ---------------------------------------------------------------- helpers
__device__ __forceinline__ ushort_t f2bf(float x) {
  uint32_t u = __float_as_uint(x);
  uint32_t r = u + 0x7FFFu + ((u >> 16) & 1u);   // round-to-nearest-even
  return (ushort_t)(r >> 16);
}

union FragU { v16bf v; f32x4 q[2]; };

// A fragment (16x32 bf16, row-major source): lane holds row m=lane&15,
// two contiguous 8-elem runs at k0 = half*8 and k0+16.
__device__ __forceinline__ v16bf load_a_frag(const ushort_t* p) {
  FragU u;
  u.q[0] = *(const f32x4*)(p);
  u.q[1] = *(const f32x4*)(p + 16);
  return u.v;
}
// B fragment (32x16 bf16, stored N-major Bt[n][k]): lane holds col n=lane&15,
// two contiguous 8-elem runs at k0 = half*8 and k0+16.
__device__ __forceinline__ v16bf load_b_frag(const ushort_t* p) {
  FragU u;
  u.q[0] = *(const f32x4*)(p);
  u.q[1] = *(const f32x4*)(p + 16);
  return u.v;
}

__device__ __forceinline__ v8f zero8() {
  v8f z;
#pragma unroll
  for (int i = 0; i < 8; ++i) z[i] = 0.0f;
  return z;
}

// Async 16-byte per-lane copy global -> LDS (gfx1250, tracked by ASYNCcnt).
__device__ __forceinline__ void async_copy16(uint32_t lds_byte_addr, const void* gaddr) {
  asm volatile("global_load_async_to_lds_b128 %0, %1, off"
               :: "v"(lds_byte_addr), "v"(gaddr) : "memory");
}
__device__ __forceinline__ void wait_async0() {
  asm volatile("s_wait_asynccnt 0x0" ::: "memory");
}

// ---------------------------------------------------------------- packing
// x[bt,d] = bf16(tok_table[idx[bt],d] + pos_table[t,d])
__global__ void k_embed(const int* __restrict__ idx, const float* __restrict__ tok,
                        const float* __restrict__ pos, ushort_t* __restrict__ x) {
  int o = blockIdx.x * 256 + threadIdx.x;
  if (o >= BT * DD) return;
  int d  = o % DD;
  int bt = o / DD;
  int t  = bt % TT;
  x[o] = f2bf(tok[(size_t)idx[bt] * DD + d] + pos[(size_t)t * DD + d]);
}

// Wqkv_t[n, k] for n in [0,3D): column n<D -> Q(h=n/HS,e=n%HS), etc.
__global__ void k_pack_qkvw(const float* __restrict__ wq, const float* __restrict__ wk,
                            const float* __restrict__ wv, ushort_t* __restrict__ dst) {
  int o = blockIdx.x * 256 + threadIdx.x;
  if (o >= D3 * DD) return;
  int k = o % DD;
  int n = o / DD;
  int which = n / DD, nn = n % DD;
  int h = nn / HS, e = nn % HS;
  const float* w = (which == 0) ? wq : ((which == 1) ? wk : wv);
  dst[o] = f2bf(w[((size_t)h * DD + k) * HS + e]);
}

// Generic transpose-pack: dst[n*K + k] = src[k*N + n], zero-pad n >= N.
__global__ void k_pack_wT(const float* __restrict__ src, ushort_t* __restrict__ dst,
                          int K, int N, int Npad) {
  int o = blockIdx.x * 256 + threadIdx.x;
  if (o >= Npad * K) return;
  int k = o % K;
  int n = o / K;
  dst[o] = (n < N) ? f2bf(src[(size_t)k * N + n]) : (ushort_t)0;
}

// Vt[((b*H+h)*HS + e)*T + s] = qkv[(b*T+s)*3D + 2D + h*HS + e]
__global__ void k_pack_vt(const ushort_t* __restrict__ qkv, ushort_t* __restrict__ vt) {
  int o = blockIdx.x * 256 + threadIdx.x;
  if (o >= BT * DD) return;
  int s = o % TT;
  int e = (o / TT) % HS;
  int h = (o / (TT * HS)) % HH;
  int b =  o / (TT * HS * HH);
  vt[o] = qkv[(size_t)(b * TT + s) * D3 + 2 * DD + h * HS + e];
}

// ---------------------------------------------------------------- GEMM
// C[M,Nstore] = A[M,K](bf16,row-major) x Bt[Npad,K](bf16,N-major) + bias.
// Workgroup tile: 256(M) x 64(N); 8 waves, each 32(M) x 64(N).
// B tile (64n x 32k) staged in LDS via async global->LDS copies, double
// buffered; A fragments streamed from global (L2-resident).
__global__ void __launch_bounds__(256)
k_gemm(const ushort_t* __restrict__ A, const ushort_t* __restrict__ Bt,
       const float* __restrict__ bias, float* __restrict__ Cf,
       ushort_t* __restrict__ Cb, int M, int Npad, int Nstore, int K,
       int ldc, int relu) {
  __shared__ alignas(16) ushort_t Btile[2][64 * 32];

  const int tid  = threadIdx.x;
  const int wave = tid >> 5;
  const int lane = tid & 31;
  const int half = lane >> 4, lm = lane & 15;

  const int ntiles = Npad >> 6;
  const int mb = (int)blockIdx.x / ntiles;     // 256-row block
  const int nt = (int)blockIdx.x - mb * ntiles;

  const int rowbase = mb * 256 + wave * 32;
  const ushort_t* arow0 = A + (size_t)(rowbase + lm) * K + half * 8;
  const ushort_t* arow1 = A + (size_t)(rowbase + 16 + lm) * K + half * 8;

  // B staging: thread t copies one 16B chunk: n_local = t/4, chunk = t%4
  const int nl = tid >> 2, ch = tid & 3;
  const ushort_t* bsrc = Bt + (size_t)(nt * 64 + nl) * K + ch * 8;
  const uint32_t ldsoff = (uint32_t)((nl * 32 + ch * 8) * 2);
  const uint32_t ldsb0 = (uint32_t)(uintptr_t)(&Btile[0][0]) + ldsoff;
  const uint32_t ldsb1 = (uint32_t)(uintptr_t)(&Btile[1][0]) + ldsoff;

  v8f acc[2][4];
#pragma unroll
  for (int g = 0; g < 2; ++g)
#pragma unroll
    for (int s = 0; s < 4; ++s) acc[g][s] = zero8();

  const int KT = K >> 5;
  // prologue: stage k-step 0 into buffer 0
  async_copy16(ldsb0, bsrc);
  wait_async0();
  __syncthreads();

  for (int kt = 0; kt < KT; ++kt) {
    const int cur = kt & 1;
    if (kt + 1 < KT)
      async_copy16(cur ? ldsb0 : ldsb1, bsrc + (kt + 1) * 32);

    v16bf af0 = load_a_frag(arow0 + kt * 32);
    v16bf af1 = load_a_frag(arow1 + kt * 32);
#pragma unroll
    for (int s = 0; s < 4; ++s) {
      v16bf bf = load_b_frag(&Btile[cur][(s * 16 + lm) * 32 + half * 8]);
      acc[0][s] = __builtin_amdgcn_wmma_f32_16x16x32_bf16(false, af0, false, bf, (short)0, acc[0][s], false, false);
      acc[1][s] = __builtin_amdgcn_wmma_f32_16x16x32_bf16(false, af1, false, bf, (short)0, acc[1][s], false, false);
    }
    wait_async0();        // next buffer filled (this thread's copy done)
    __syncthreads();      // all waves done reading cur + done staging next
  }

#pragma unroll
  for (int g = 0; g < 2; ++g) {
#pragma unroll
    for (int r = 0; r < 8; ++r) {
      int m = rowbase + g * 16 + half * 8 + r;
      float vals[4] = { acc[g][0][r], acc[g][1][r], acc[g][2][r], acc[g][3][r] };
#pragma unroll
      for (int s = 0; s < 4; ++s) {
        int n = nt * 64 + s * 16 + lm;
        if (n < Nstore) {
          float v = vals[s] + (bias ? bias[n] : 0.0f);
          if (relu) v = fmaxf(v, 0.0f);
          if (Cf) Cf[(size_t)m * ldc + n] = v;
          if (Cb) Cb[(size_t)m * ldc + n] = f2bf(v);
        }
      }
    }
  }
}

// ---------------------------------------------------------------- attention
// One wave per (b, h, 16-row tile). S = Q K^T (causal, scaled D^-0.5),
// softmax in C-fragments, P -> LDS (layout conversion), O = P V.
__global__ void __launch_bounds__(32)
k_attn(const ushort_t* __restrict__ qkv, const ushort_t* __restrict__ vt,
       ushort_t* __restrict__ out) {
  __shared__ alignas(16) ushort_t P[16 * TT];   // 16 rows x 256 keys, bf16

  int itile = blockIdx.x & 15;
  int h = (blockIdx.x >> 4) & (HH - 1);
  int b = blockIdx.x >> 6;
  int lane = threadIdx.x;
  int half = lane >> 4, lm = lane & 15;

  // Q A-fragments for the 3 K-steps (HS = 96 = 3*32)
  const ushort_t* qbase =
      qkv + (size_t)(b * TT + itile * 16 + lm) * D3 + h * HS + half * 8;
  v16bf qf0 = load_a_frag(qbase);
  v16bf qf1 = load_a_frag(qbase + 32);
  v16bf qf2 = load_a_frag(qbase + 64);

  const ushort_t* kbase = qkv + (size_t)(b * TT) * D3 + DD + h * HS + half * 8;

  v8f acc[16];
#pragma unroll
  for (int j = 0; j < 16; ++j) acc[j] = zero8();

#pragma unroll
  for (int j = 0; j < 16; ++j) {
    if (j <= itile) {                          // uniform branch, EXEC stays full
      const ushort_t* kb = kbase + (size_t)(j * 16 + lm) * D3;
      v16bf k0 = load_b_frag(kb);
      v16bf k1 = load_b_frag(kb + 32);
      v16bf k2 = load_b_frag(kb + 64);
      acc[j] = __builtin_amdgcn_wmma_f32_16x16x32_bf16(false, qf0, false, k0, (short)0, acc[j], false, false);
      acc[j] = __builtin_amdgcn_wmma_f32_16x16x32_bf16(false, qf1, false, k1, (short)0, acc[j], false, false);
      acc[j] = __builtin_amdgcn_wmma_f32_16x16x32_bf16(false, qf2, false, k2, (short)0, acc[j], false, false);
    }
  }

  const float scale = 0.051031036307982884f;   // 384^-0.5 (reference uses D)
  const float NEG = -3.4e38f;
  float rmax[8], rsum[8];
#pragma unroll
  for (int r = 0; r < 8; ++r) rmax[r] = NEG;

  // scale + causal mask + row max
#pragma unroll
  for (int j = 0; j < 16; ++j) {
    if (j <= itile) {
      int tk = j * 16 + lm;
#pragma unroll
      for (int r = 0; r < 8; ++r) {
        int tq = itile * 16 + half * 8 + r;
        float s = acc[j][r] * scale;
        if (tk > tq) s = NEG;
        acc[j][r] = s;
        rmax[r] = fmaxf(rmax[r], s);
      }
    }
  }
#pragma unroll
  for (int off = 1; off < 16; off <<= 1)
#pragma unroll
    for (int r = 0; r < 8; ++r)
      rmax[r] = fmaxf(rmax[r], __shfl_xor(rmax[r], off, 32));

#pragma unroll
  for (int r = 0; r < 8; ++r) rsum[r] = 0.0f;
#pragma unroll
  for (int j = 0; j < 16; ++j) {
    if (j <= itile) {
#pragma unroll
      for (int r = 0; r < 8; ++r) {
        float p = __expf(acc[j][r] - rmax[r]);  // masked -> 0
        acc[j][r] = p;
        rsum[r] += p;
      }
    }
  }
#pragma unroll
  for (int off = 1; off < 16; off <<= 1)
#pragma unroll
    for (int r = 0; r < 8; ++r)
      rsum[r] += __shfl_xor(rsum[r], off, 32);

  float rinv[8];
#pragma unroll
  for (int r = 0; r < 8; ++r) rinv[r] = 1.0f / rsum[r];

  // P (C-layout) -> LDS row-major bf16 [16][256]
#pragma unroll
  for (int j = 0; j < 16; ++j) {
#pragma unroll
    for (int r = 0; r < 8; ++r) {
      ushort_t pv = (j <= itile) ? f2bf(acc[j][r] * rinv[r]) : (ushort_t)0;
      P[(half * 8 + r) * TT + j * 16 + lm] = pv;
    }
  }
  __syncthreads();

  // O = P @ V  (HS = 96 -> 6 N-tiles; K = 256 -> 8 steps)
#pragma unroll
  for (int n6 = 0; n6 < 6; ++n6) {
    v8f o8 = zero8();
    const ushort_t* vb =
        vt + (size_t)((b * HH + h) * HS + n6 * 16 + lm) * TT + half * 8;
#pragma unroll
    for (int kk = 0; kk < 8; ++kk) {
      v16bf pf = load_a_frag(&P[lm * TT + kk * 32 + half * 8]);
      v16bf vf = load_b_frag(vb + kk * 32);
      o8 = __builtin_amdgcn_wmma_f32_16x16x32_bf16(false, pf, false, vf, (short)0, o8, false, false);
    }
#pragma unroll
    for (int r = 0; r < 8; ++r) {
      int t = itile * 16 + half * 8 + r;
      out[(size_t)(b * TT + t) * DD + h * HS + n6 * 16 + lm] = f2bf(o8[r]);
    }
  }
}

// ---------------------------------------------------------------- loss
__global__ void k_rowloss(const float* __restrict__ logits, const int* __restrict__ tgt,
                          float* __restrict__ rowloss) {
  int bt = blockIdx.x * 256 + threadIdx.x;
  if (bt >= BT) return;
  const float* row = logits + (size_t)bt * VV;
  float mx = -3.4e38f;
  for (int c = 0; c < VV; ++c) mx = fmaxf(mx, row[c]);
  float s = 0.0f;
  for (int c = 0; c < VV; ++c) s += __expf(row[c] - mx);
  float lse = mx + __logf(s);
  rowloss[bt] = lse - row[tgt[bt]];            // = -log p[target]
}

__global__ void k_redloss(const float* __restrict__ rowloss, float* __restrict__ out) {
  __shared__ float sm[256];
  int tid = threadIdx.x;
  float s = 0.0f;
  for (int i = tid; i < BT; i += 256) s += rowloss[i];
  sm[tid] = s;
  __syncthreads();
  for (int o = 128; o > 0; o >>= 1) {
    if (tid < o) sm[tid] += sm[tid + o];
    __syncthreads();
  }
  if (tid == 0) out[0] = sm[0] / (float)BT;
}

// ---------------------------------------------------------------- launcher
extern "C" void kernel_launch(void* const* d_in, const int* in_sizes, int n_in,
                              void* d_out, int out_size, void* d_ws, size_t ws_size,
                              hipStream_t stream) {
  (void)in_sizes; (void)n_in; (void)out_size; (void)ws_size;
  const int*   idx   = (const int*)d_in[0];
  const int*   tgt   = (const int*)d_in[1];
  const float* tok   = (const float*)d_in[2];
  const float* pos   = (const float*)d_in[3];
  const float* wq    = (const float*)d_in[4];
  const float* wk    = (const float*)d_in[5];
  const float* wv    = (const float*)d_in[6];
  const float* projw = (const float*)d_in[7];
  const float* projb = (const float*)d_in[8];
  const float* w1    = (const float*)d_in[9];
  const float* b1    = (const float*)d_in[10];
  const float* w2    = (const float*)d_in[11];
  const float* b2    = (const float*)d_in[12];
  const float* lmw   = (const float*)d_in[13];
  const float* lmb   = (const float*)d_in[14];
  float* out = (float*)d_out;

  char* ws = (char*)d_ws;
  size_t off = 0;
  auto alloc = [&](size_t bytes) -> void* {
    void* p = ws + off;
    off = (off + bytes + 255) & ~(size_t)255;
    return p;
  };

  ushort_t* xbuf   = (ushort_t*)alloc((size_t)BT * DD * 2);   // x / attn / ffn2-out
  ushort_t* qkvbuf = (ushort_t*)alloc((size_t)BT * D3 * 2);   // qkv
  ushort_t* vtbuf  = (ushort_t*)alloc((size_t)BT * DD * 2);   // V^T / proj-out
  ushort_t* midbuf = (ushort_t*)alloc((size_t)BT * DF * 2);   // ffn hidden
  ushort_t* wqkvT  = (ushort_t*)alloc((size_t)D3 * DD * 2);
  ushort_t* projT  = (ushort_t*)alloc((size_t)DD * DD * 2);
  ushort_t* w1T    = (ushort_t*)alloc((size_t)DF * DD * 2);
  ushort_t* w2T    = (ushort_t*)alloc((size_t)DD * DF * 2);
  ushort_t* lmT    = (ushort_t*)alloc((size_t)VPAD * DD * 2);
  float*    rloss  = (float*)alloc((size_t)BT * 4);

  ushort_t* attn = xbuf;     // x dead after qkv GEMM
  ushort_t* proj = vtbuf;    // V^T dead after attention
  ushort_t* ffno = xbuf;     // attn dead after proj GEMM

  // packing / embedding
  k_embed<<<(BT * DD + 255) / 256, 256, 0, stream>>>(idx, tok, pos, xbuf);
  k_pack_qkvw<<<(D3 * DD + 255) / 256, 256, 0, stream>>>(wq, wk, wv, wqkvT);
  k_pack_wT<<<(DD * DD + 255) / 256, 256, 0, stream>>>(projw, projT, DD, DD, DD);
  k_pack_wT<<<(DF * DD + 255) / 256, 256, 0, stream>>>(w1, w1T, DD, DF, DF);
  k_pack_wT<<<(DD * DF + 255) / 256, 256, 0, stream>>>(w2, w2T, DF, DD, DD);
  k_pack_wT<<<(VPAD * DD + 255) / 256, 256, 0, stream>>>(lmw, lmT, DD, VV, VPAD);

  auto gemm = [&](const ushort_t* A, const ushort_t* Bt, const float* bias,
                  float* Cf, ushort_t* Cb, int M, int Npad, int Nstore, int K,
                  int ldc, int relu) {
    int blocks = (M >> 8) * (Npad >> 6);   // 256-row x 64-col workgroup tiles
    k_gemm<<<blocks, 256, 0, stream>>>(A, Bt, bias, Cf, Cb, M, Npad, Nstore, K, ldc, relu);
  };

  // QKV: [BT,384] x [384,1152]
  gemm(xbuf, wqkvT, nullptr, nullptr, qkvbuf, BT, D3, D3, DD, D3, 0);
  k_pack_vt<<<(BT * DD + 255) / 256, 256, 0, stream>>>(qkvbuf, vtbuf);

  // attention: one wave per (b, h, 16-row tile)
  k_attn<<<BB * HH * (TT / 16), 32, 0, stream>>>(qkvbuf, vtbuf, attn);

  // proj, FFN, LM head
  gemm(attn,   projT, projb, nullptr, proj,   BT, DD, DD, DD, DD, 0);
  gemm(proj,   w1T,   b1,    nullptr, midbuf, BT, DF, DF, DD, DF, 1);
  gemm(midbuf, w2T,   b2,    nullptr, ffno,   BT, DD, DD, DF, DD, 0);
  gemm(ffno,   lmT,   lmb,   out,     nullptr, BT, VPAD, VV, DD, VV, 0);

  // loss
  k_rowloss<<<(BT + 255) / 256, 256, 0, stream>>>(out, tgt, rloss);
  k_redloss<<<1, 256, 0, stream>>>(rloss, out + (size_t)BT * VV);
}
